// _HModule_54391465837134
// MI455X (gfx1250) — compile-verified, matrix operand order checked by reading, not run on hardware
//
#include <hip/hip_runtime.h>
#include <hip/hip_bf16.h>
#include <math.h>

// Problem constants
#define D_   1024
#define H_   16
#define HD_  64
#define C_   50000
#define NTILES_C 3125          // C_/16
#define NCHUNK 25
#define CHUNK  2000            // NCHUNK*CHUNK == C_

typedef __attribute__((ext_vector_type(16))) __bf16 v16bf;
typedef __attribute__((ext_vector_type(8)))  float  v8f;

// ---- workspace layout (float offsets), total ~8.2 MB ----
#define WS_Q      0                       // 1024
#define WS_QK     1024                    // 16*1024
#define WS_SBIAS  (WS_QK + H_*D_)         // 16
#define WS_SSUM   (WS_SBIAS + 16)         // 16
#define WS_SMAX   (WS_SSUM + 16)          // 16
#define WS_SCORES (WS_SMAX + 16 + 16)     // C_*16  (padded to 8-float align)
#define WS_WEXP   (WS_SCORES + C_*16)     // C_*16
#define WS_UPART  (WS_WEXP + C_*16)       // NCHUNK*16*1024
#define WS_U      (WS_UPART + NCHUNK*H_*D_) // 16*1024
#define WS_Z2     (WS_U + H_*D_)          // 2048
#define WS_H1     (WS_Z2 + 2*D_)          // 1024

// ---------------- K1: Q = z_prev @ Wq + bq ----------------
__global__ void k_q(const float* __restrict__ z_prev, const float* __restrict__ Wq,
                    const float* __restrict__ bq, float* __restrict__ ws) {
    int j = blockIdx.x * 256 + threadIdx.x;          // 0..1023
    float acc = bq[j];
    for (int d = 0; d < D_; ++d) acc = fmaf(z_prev[d], Wq[d * D_ + j], acc);
    ws[WS_Q + j] = acc;
}

// ---- K2: qk[h,d] = (Wk[d, h*64:+64] . Q[h]) / 8 ; sbias[h] = (bk_h . Q[h]) / 8 ----
__global__ void k_qk(const float* __restrict__ Wk, const float* __restrict__ bk,
                     float* __restrict__ ws) {
    int t = blockIdx.x * 256 + threadIdx.x;          // 0..16383 ; t = h*1024 + d
    int h = t >> 10, d = t & 1023;
    const float* q    = ws + WS_Q + h * HD_;
    const float* wrow = Wk + (size_t)d * D_ + h * HD_;
    float acc = 0.f;
    #pragma unroll 8
    for (int i = 0; i < HD_; ++i) acc = fmaf(wrow[i], q[i], acc);
    ws[WS_QK + t] = acc * 0.125f;                    // 1/sqrt(64)
    if (t < H_) {
        const float* qq = ws + WS_Q + t * HD_;
        float s = 0.f;
        for (int i = 0; i < HD_; ++i) s = fmaf(bk[t * HD_ + i], qq[i], s);
        ws[WS_SBIAS + t] = s * 0.125f;
    }
}

// bf16 A/B fragment packing per ISA 7.12.2: element (2v+e) of the 16-vector holds
// K = (v>>2)*16 + (lane>>4)*8 + (v&3)*2 + e  => two runs of 8 consecutive K per lane.
__device__ __forceinline__ void pack8(v16bf& f, int base, const float* p) {
    const float4* p4 = (const float4*)p;             // 32B-aligned here
    float4 a = p4[0], b = p4[1];
    f[base+0]=(__bf16)a.x; f[base+1]=(__bf16)a.y; f[base+2]=(__bf16)a.z; f[base+3]=(__bf16)a.w;
    f[base+4]=(__bf16)b.x; f[base+5]=(__bf16)b.y; f[base+6]=(__bf16)b.z; f[base+7]=(__bf16)b.w;
}

// ---- K3 (WMMA pass A): scores[c,h] = nodes[c,:] . qk[h,:] + sbias[h] ----
__global__ void k_scores(const float* __restrict__ nodes, float* __restrict__ ws) {
    int wave = threadIdx.x >> 5, lane = threadIdx.x & 31;
    int tile = blockIdx.x * 8 + wave;
    if (tile >= NTILES_C) return;                    // wave-uniform; EXEC stays full
    int half = lane >> 4, mn = lane & 15;
    const float* arow   = nodes + (size_t)(tile * 16 + mn) * D_;  // A: M = lane&15
    const float* qkbase = ws + WS_QK + mn * D_;                   // B: N(head) = lane&15
    v8f acc = {};
    for (int k0 = 0; k0 < D_; k0 += 32) {
        v16bf a, b;
        #pragma unroll
        for (int g = 0; g < 2; ++g) {
            int off = k0 + g * 16 + half * 8;
            pack8(a, g * 8, arow + off);
            pack8(b, g * 8, qkbase + off);
        }
        acc = __builtin_amdgcn_wmma_f32_16x16x32_bf16(false, a, false, b,
                                                      (short)0, acc, false, false);
    }
    float sb = ws[WS_SBIAS + mn];
    float* sc = ws + WS_SCORES;
    int c_base = tile * 16 + half * 8;               // C/D: M = half*8 + j, N = lane&15
    #pragma unroll
    for (int j = 0; j < 8; ++j) sc[(c_base + j) * 16 + mn] = acc[j] + sb;
}

// ---- K4: per-head softmax stats + unnormalized exp weights ----
__global__ void k_softmax(float* __restrict__ ws) {
    __shared__ float red[256];
    int h = blockIdx.x, tid = threadIdx.x;
    const float* sc = ws + WS_SCORES;
    float m = -3.4e38f;
    for (int c = tid; c < C_; c += 256) m = fmaxf(m, sc[c * 16 + h]);
    red[tid] = m; __syncthreads();
    for (int s = 128; s > 0; s >>= 1) { if (tid < s) red[tid] = fmaxf(red[tid], red[tid + s]); __syncthreads(); }
    m = red[0]; __syncthreads();
    float* we = ws + WS_WEXP;
    float sum = 0.f;
    for (int c = tid; c < C_; c += 256) {
        float e = expf(sc[c * 16 + h] - m);
        we[c * 16 + h] = e; sum += e;
    }
    red[tid] = sum; __syncthreads();
    for (int s = 128; s > 0; s >>= 1) { if (tid < s) red[tid] += red[tid + s]; __syncthreads(); }
    if (tid == 0) { ws[WS_SSUM + h] = red[0]; ws[WS_SMAX + h] = m; }
}

// ---- K5 (WMMA pass B): u_part[chunk] = wexp[16 x CHUNK] @ nodes[CHUNK x 1024] ----
__global__ void k_pool_gemm(const float* __restrict__ nodes, float* __restrict__ ws) {
    int wave = threadIdx.x >> 5, lane = threadIdx.x & 31;
    int ngrp = blockIdx.x & 7, chunk = blockIdx.x >> 3;
    int n0 = (ngrp * 8 + wave) * 16;
    int c0 = chunk * CHUNK;
    int half = lane >> 4, mn = lane & 15;
    const float* we = ws + WS_WEXP;
    v8f acc = {};
    for (int k0 = 0; k0 < CHUNK; k0 += 32) {         // 63 iters, last half zero-filled
        v16bf a, b;
        #pragma unroll
        for (int g = 0; g < 2; ++g) {
            int kb = k0 + g * 16 + half * 8;
            #pragma unroll
            for (int j = 0; j < 8; ++j) {
                int kk = kb + j, c = c0 + kk;
                a[g * 8 + j] = (kk < CHUNK) ? (__bf16)we[c * 16 + mn] : (__bf16)0.0f;
                b[g * 8 + j] = (c < C_) ? (__bf16)nodes[(size_t)c * D_ + n0 + mn] : (__bf16)0.0f;
            }
        }
        acc = __builtin_amdgcn_wmma_f32_16x16x32_bf16(false, a, false, b,
                                                      (short)0, acc, false, false);
    }
    float* up = ws + WS_UPART + chunk * (H_ * D_);
    #pragma unroll
    for (int j = 0; j < 8; ++j) up[(half * 8 + j) * D_ + n0 + mn] = acc[j];
}

// ---- K5b: deterministic fixed-order reduction over chunks ----
__global__ void k_ureduce(float* __restrict__ ws) {
    int t = blockIdx.x * 256 + threadIdx.x;          // 0..16383
    float s = 0.f;
    const float* up = ws + WS_UPART;
    for (int c = 0; c < NCHUNK; ++c) s += up[c * (H_ * D_) + t];
    ws[WS_U + t] = s;
}

// ---- K6: pooled = (u/ssum) @ Wv-slice + bv ; build z2 = [pooled, z_prev] ----
__global__ void k_pooled(const float* __restrict__ Wv, const float* __restrict__ bv,
                         const float* __restrict__ z_prev, float* __restrict__ ws) {
    int t = blockIdx.x * 256 + threadIdx.x;          // 0..1023
    int h = t >> 6;
    float inv = 1.0f / ws[WS_SSUM + h];
    const float* u = ws + WS_U + h * D_;
    float acc = 0.f;
    for (int d = 0; d < D_; ++d) acc = fmaf(u[d], Wv[(size_t)d * D_ + t], acc);
    ws[WS_Z2 + t]      = acc * inv + bv[t];
    ws[WS_Z2 + D_ + t] = z_prev[t];
}

// ---- K7: h1 = GELU_exact(z2 @ W1 + b1) ----
__global__ void k_mlp1(const float* __restrict__ W1, const float* __restrict__ b1,
                       float* __restrict__ ws) {
    int j = blockIdx.x * 256 + threadIdx.x;
    const float* z2 = ws + WS_Z2;
    float acc = b1[j];
    for (int i = 0; i < 2 * D_; ++i) acc = fmaf(z2[i], W1[(size_t)i * D_ + j], acc);
    ws[WS_H1 + j] = 0.5f * acc * (1.0f + erff(acc * 0.70710678118654752f));
}

// ---- K8: y = h1 @ W2 + b2 ; out = scale * y * rsqrt(mean(y^2)+eps) ----
__global__ void k_out(const float* __restrict__ W2, const float* __restrict__ b2,
                      const float* __restrict__ scale, float* __restrict__ ws,
                      float* __restrict__ out) {
    __shared__ float red[1024];
    int j = threadIdx.x;
    const float* h1 = ws + WS_H1;
    float acc = b2[j];
    for (int i = 0; i < D_; ++i) acc = fmaf(h1[i], W2[(size_t)i * D_ + j], acc);
    red[j] = acc * acc;
    __syncthreads();
    for (int s = 512; s > 0; s >>= 1) { if (j < s) red[j] += red[j + s]; __syncthreads(); }
    float r = rsqrtf(red[0] * (1.0f / 1024.0f) + 1e-6f);
    out[j] = scale[j] * acc * r;
}

extern "C" void kernel_launch(void* const* d_in, const int* in_sizes, int n_in,
                              void* d_out, int out_size, void* d_ws, size_t ws_size,
                              hipStream_t stream) {
    const float* nodes  = (const float*)d_in[0];
    const float* z_prev = (const float*)d_in[1];
    const float* Wq     = (const float*)d_in[2];
    const float* bq     = (const float*)d_in[3];
    const float* Wk     = (const float*)d_in[4];
    const float* bk     = (const float*)d_in[5];
    const float* Wv     = (const float*)d_in[6];
    const float* bv     = (const float*)d_in[7];
    const float* W1     = (const float*)d_in[8];
    const float* b1     = (const float*)d_in[9];
    const float* W2     = (const float*)d_in[10];
    const float* b2     = (const float*)d_in[11];
    const float* scale  = (const float*)d_in[12];
    float* ws  = (float*)d_ws;
    float* out = (float*)d_out;

    k_q        <<<4,   256, 0, stream>>>(z_prev, Wq, bq, ws);
    k_qk       <<<64,  256, 0, stream>>>(Wk, bk, ws);
    k_scores   <<<391, 256, 0, stream>>>(nodes, ws);   // 3125 wave-tiles, 8 waves/block
    k_softmax  <<<16,  256, 0, stream>>>(ws);
    k_pool_gemm<<<200, 256, 0, stream>>>(nodes, ws);   // 25 chunks x 8 col-groups
    k_ureduce  <<<64,  256, 0, stream>>>(ws);
    k_pooled   <<<4,   256, 0, stream>>>(Wv, bv, z_prev, ws);
    k_mlp1     <<<4,   256, 0, stream>>>(W1, b1, ws);
    k_out      <<<1,  1024, 0, stream>>>(W2, b2, scale, ws, out);
}
